// EDVRFeatureExtractor_8770323219144
// MI455X (gfx1250) — matrix operand order repaired and checked by reading, not compiled, not run on hardware
//
#include <hip/hip_runtime.h>

// ---------------------------------------------------------------------------
// EDVR feature extractor for gfx1250 (MI455X).
// All convolutions (3x3 and 1x1, incl. deformable-conv aggregation as K=576
// 1x1) run through one implicit-GEMM kernel built on
// v_wmma_f32_16x16x32_f16 (wave32, f32 accumulate), 2 WMMAs per wave per
// K-step (16 cout x 32 px tile). 1x1 interior tiles stage input via
// global_load_async_to_lds_b128 (async-DMA path, ASYNCcnt).
// ---------------------------------------------------------------------------

typedef _Float16 half_t;
typedef __attribute__((ext_vector_type(16))) _Float16 v16h;
typedef __attribute__((ext_vector_type(8)))  _Float16 v8h;
typedef __attribute__((ext_vector_type(8)))  float    v8f;
typedef int v4i_t __attribute__((vector_size(16)));

#define ACT_NONE  0
#define ACT_LRELU 1
#define ACT_RELU  2

#ifdef __has_builtin
#if __has_builtin(__builtin_amdgcn_global_load_async_to_lds_b128)
#define USE_ASYNC_LDS 1
#endif
#endif
#ifndef USE_ASYNC_LDS
#define USE_ASYNC_LDS 0
#endif

#if USE_ASYNC_LDS
__device__ __forceinline__ void async_ld_b128(const half_t* g, half_t* l) {
  __builtin_amdgcn_global_load_async_to_lds_b128(
      (__attribute__((address_space(1))) v4i_t*)g,
      (__attribute__((address_space(3))) v4i_t*)l, 0, 0);
}
__device__ __forceinline__ void wait_async0() {
#if __has_builtin(__builtin_amdgcn_s_wait_asynccnt)
  __builtin_amdgcn_s_wait_asynccnt(0);
#else
  asm volatile("s_wait_asynccnt 0" ::: "memory");
#endif
}
#endif

__device__ __forceinline__ float dsigmoid(float x) { return 1.0f / (1.0f + __expf(-x)); }

// ---------------------------------------------------------------------------
// Implicit-GEMM conv:  out[co, oy, ox] = act( sum_k W[co,k] * im2col[k, pix] + b[co] )
// K = Cin*ks*ks, weights pre-packed f16 [CoutPad rows][Kpad cols].
// Block = 128 threads (4 waves). Wave: 16 cout x 32 px (2 accumulators),
// K in steps of 32; shared im2col tile [32 px][32 k] built cooperatively.
// ---------------------------------------------------------------------------
template <int KS, int STRIDE>
__global__ __launch_bounds__(128) void conv_wmma_kernel(
    const half_t* __restrict__ in, const half_t* __restrict__ wgt,
    const float* __restrict__ bias, half_t* __restrict__ out,
    int Cin, int H, int W, int Cout, int Ho, int Wo,
    int act, int zShift, int zMask,
    long inImgStride, long outImgStride)
{
  constexpr int taps = KS * KS;
  constexpr int rows = (KS == 3) ? 3 : 1;
  constexpr int PW   = (KS == 3) ? (32 * STRIDE + 2) : 32;
  // max Cin per variant: 3x3 s1 -> 128, 3x3 s2 -> 64, 1x1 -> 576
  constexpr int PATCH_MAX = (KS == 1) ? (576 * 32)
                                      : ((STRIDE == 1) ? (128 * 3 * 34) : (64 * 3 * 66));

  __shared__ __align__(16) half_t s_patch[PATCH_MAX];
  __shared__ __align__(16) half_t s_imcol[32 * 40];  // [px][k], row stride 40 halves

  const int tid  = threadIdx.x;
  const int wave = tid >> 5;
  const int lane = tid & 31;
  const int hlf  = lane >> 4;   // 0 or 1
  const int l16  = lane & 15;

  const int oy      = blockIdx.y;
  const int img     = blockIdx.z >> zShift;
  const int coutBlk = blockIdx.z & zMask;
  const int pxbase  = blockIdx.x * 32;

  const int K       = Cin * taps;
  const int Kpad    = (K + 31) & ~31;
  const int CoutPad = (Cout + 15) & ~15;

  const half_t* inImg = in + (long)img * inImgStride;

  // ---- stage input patch into LDS ----
  bool staged = false;
#if USE_ASYNC_LDS
  if (KS == 1 && pxbase + 32 <= W) {
    // interior 1x1 tile: each channel row is 32 contiguous halves = 4 x b128
    const half_t* rowp = inImg + (long)oy * W + pxbase;
    for (int u = tid; u < Cin * 4; u += 128) {
      int ci = u >> 2, q = u & 3;
      async_ld_b128(rowp + (long)ci * H * W + q * 8, &s_patch[ci * 32 + q * 8]);
    }
    wait_async0();
    staged = true;
  }
#endif
  if (!staged) {
    const int patchN = Cin * rows * PW;
    for (int i = tid; i < patchN; i += 128) {
      int ci  = i / (rows * PW);
      int rem = i - ci * (rows * PW);
      int r   = rem / PW;
      int c   = rem - r * PW;
      int iy, ix;
      if (KS == 3) { iy = oy * STRIDE - 1 + r; ix = pxbase * STRIDE - 1 + c; }
      else         { iy = oy;                  ix = pxbase + c; }
      half_t v = (half_t)0.0f;
      if (iy >= 0 && iy < H && ix >= 0 && ix < W)
        v = inImg[((long)ci * H + iy) * W + ix];
      s_patch[i] = v;
    }
  }
  __syncthreads();

  const int  waveRow    = coutBlk * 64 + wave * 16;
  const bool waveActive = (waveRow < CoutPad);
  const half_t* wrow    = wgt + (long)(waveRow + l16) * Kpad;

  v8f acc0 = {0.f, 0.f, 0.f, 0.f, 0.f, 0.f, 0.f, 0.f};
  v8f acc1 = {0.f, 0.f, 0.f, 0.f, 0.f, 0.f, 0.f, 0.f};

  for (int k0 = 0; k0 < Kpad; k0 += 32) {
    // ---- cooperative im2col tile build: [32 px][32 k] ----
    for (int i = tid; i < 1024; i += 128) {
      int px = i >> 5;
      int kk = i & 31;
      int k  = k0 + kk;
      half_t v = (half_t)0.0f;
      if (k < K) {
        int ci  = k / taps;            // taps is a compile-time constant
        int tap = k - ci * taps;
        if (KS == 3) {
          int r = tap / 3, c = tap - r * 3;
          v = s_patch[(ci * 3 + r) * PW + px * STRIDE + c];
        } else {
          v = s_patch[ci * 32 + px];
        }
      }
      s_imcol[px * 40 + kk] = v;
    }
    __syncthreads();

    if (waveActive) {
      union { v16h v; v8h h[2]; } A, B0, B1;
      // A (weights) 16x32: lanes 0-15 hold K k0..k0+7 / k0+16..23 of row M=l16,
      //                    lanes 16-31 hold K k0+8..15 / k0+24..31.
      A.h[0] = *(const v8h*)(wrow + k0 + hlf * 8);
      A.h[1] = *(const v8h*)(wrow + k0 + 16 + hlf * 8);
      // B (im2col) 32x16 per tile: lane holds column N, same K split as A.
      const half_t* b0 = &s_imcol[l16 * 40 + hlf * 8];
      const half_t* b1 = &s_imcol[(16 + l16) * 40 + hlf * 8];
      B0.h[0] = *(const v8h*)(b0);
      B0.h[1] = *(const v8h*)(b0 + 16);
      B1.h[0] = *(const v8h*)(b1);
      B1.h[1] = *(const v8h*)(b1 + 16);
      if (k0 + 32 < Kpad)
        __builtin_prefetch((const void*)(wrow + k0 + 32), 0, 1);
      acc0 = __builtin_amdgcn_wmma_f32_16x16x32_f16(false, A.v, false, B0.v,
                                                    (short)0, acc0, false, false);
      acc1 = __builtin_amdgcn_wmma_f32_16x16x32_f16(false, A.v, false, B1.v,
                                                    (short)0, acc1, false, false);
    }
    __syncthreads();
  }

  if (!waveActive) return;
  half_t* outImg = out + (long)img * outImgStride;
  const int px0 = pxbase + l16;
  const int px1 = px0 + 16;
#pragma unroll
  for (int r = 0; r < 8; ++r) {
    int co = waveRow + r + hlf * 8;   // C/D layout: M = r + 8*(lane>=16), N = lane%16
    if (co < Cout) {
      float b = bias[co];
      long rowOff = (long)co * Ho + oy;
      if (px0 < Wo) {
        float v = acc0[r] + b;
        if (act == ACT_LRELU)     v = (v >= 0.f) ? v : 0.1f * v;
        else if (act == ACT_RELU) v = (v > 0.f) ? v : 0.f;
        outImg[rowOff * Wo + px0] = (half_t)v;
      }
      if (px1 < Wo) {
        float v = acc1[r] + b;
        if (act == ACT_LRELU)     v = (v >= 0.f) ? v : 0.1f * v;
        else if (act == ACT_RELU) v = (v > 0.f) ? v : 0.f;
        outImg[rowOff * Wo + px1] = (half_t)v;
      }
    }
  }
}

// ---------------------------------------------------------------------------
// Utility kernels
// ---------------------------------------------------------------------------
__global__ void pack_weight_kernel(const float* __restrict__ w, half_t* __restrict__ o,
                                   int Cout, int K, int Kpad, int total)
{
  int i = blockIdx.x * 256 + threadIdx.x;
  if (i >= total) return;
  int r = i / Kpad;
  int k = i - r * Kpad;
  float v = (r < Cout && k < K) ? w[(long)r * K + k] : 0.f;
  o[i] = (half_t)v;
}

__global__ void cvt_f32_f16_kernel(const float* __restrict__ in, half_t* __restrict__ out, int n)
{
  int i = blockIdx.x * 256 + threadIdx.x;
  if (i < n) out[i] = (half_t)in[i];
}

__global__ void add_f16_kernel(half_t* __restrict__ a, const half_t* __restrict__ b, int n)
{
  int i = blockIdx.x * 256 + threadIdx.x;
  if (i < n) a[i] = (half_t)((float)a[i] + (float)b[i]);
}

// concat helper: dst[img, dstOff+c, :] = src[simg, c, :]; centerMode -> simg = center of 5-group
__global__ void copy_ch_kernel(const half_t* __restrict__ src, half_t* __restrict__ dst,
                               int C, int HW, int nImg, int dstC, int dstOff, int centerMode)
{
  int i = blockIdx.x * 256 + threadIdx.x;
  int total = nImg * C * HW;
  if (i >= total) return;
  int img = i / (C * HW);
  int rem = i - img * C * HW;
  int c   = rem / HW;
  int p   = rem - c * HW;
  int simg = centerMode ? ((img / 5) * 5 + 2) : img;
  dst[((long)img * dstC + dstOff + c) * HW + p] = src[((long)simg * C + c) * HW + p];
}

// bilinear 2x upsample (half-pixel, edge clamp), optional scalar multiply
__global__ void up2_kernel(const half_t* __restrict__ in, half_t* __restrict__ out,
                           int C, int H, int W, int nImg, float mul)
{
  int Ho = 2 * H, Wo = 2 * W;
  int total = nImg * C * Ho * Wo;
  int i = blockIdx.x * 256 + threadIdx.x;
  if (i >= total) return;
  int p    = i % (Ho * Wo);
  int rest = i / (Ho * Wo);
  int c    = rest % C;
  int img  = rest / C;
  int oy = p / Wo, ox = p - oy * Wo;
  float fy = oy * 0.5f - 0.25f;
  float fx = ox * 0.5f - 0.25f;
  int y0 = (int)floorf(fy);
  int x0 = (int)floorf(fx);
  float ty = fy - (float)y0;
  float tx = fx - (float)x0;
  int y1 = y0 + 1, x1 = x0 + 1;
  int y0c = y0 < 0 ? 0 : (y0 > H - 1 ? H - 1 : y0);
  int y1c = y1 < 0 ? 0 : (y1 > H - 1 ? H - 1 : y1);
  int x0c = x0 < 0 ? 0 : (x0 > W - 1 ? W - 1 : x0);
  int x1c = x1 < 0 ? 0 : (x1 > W - 1 ? W - 1 : x1);
  const half_t* pc = in + ((long)img * C + c) * H * W;
  float v00 = (float)pc[y0c * W + x0c];
  float v01 = (float)pc[y0c * W + x1c];
  float v10 = (float)pc[y1c * W + x0c];
  float v11 = (float)pc[y1c * W + x1c];
  float v = (1.f - ty) * ((1.f - tx) * v00 + tx * v01) + ty * ((1.f - tx) * v10 + tx * v11);
  out[((long)img * C + c) * Ho * Wo + p] = (half_t)(v * mul);
}

// 3x3 stride-2 pad-1 max+avg pool, concatenated: out channels [0,C)=max, [C,2C)=avg
__global__ void pool_dual_kernel(const half_t* __restrict__ in, half_t* __restrict__ out,
                                 int C, int H, int W, int nImg)
{
  int Ho = H / 2, Wo = W / 2;
  int total = nImg * C * Ho * Wo;
  int i = blockIdx.x * 256 + threadIdx.x;
  if (i >= total) return;
  int p    = i % (Ho * Wo);
  int rest = i / (Ho * Wo);
  int c    = rest % C;
  int img  = rest / C;
  int oy = p / Wo, ox = p - oy * Wo;
  const half_t* pc = in + ((long)img * C + c) * H * W;
  float mx = -1e30f, sm = 0.f;
  for (int dy = 0; dy < 3; ++dy)
    for (int dx = 0; dx < 3; ++dx) {
      int y = 2 * oy - 1 + dy, x = 2 * ox - 1 + dx;
      if (y >= 0 && y < H && x >= 0 && x < W) {
        float v = (float)pc[y * W + x];
        mx = fmaxf(mx, v);
        sm += v;
      }
    }
  half_t* op = out + (long)img * 2 * C * Ho * Wo;
  op[(long)c * Ho * Wo + p]       = (half_t)mx;
  op[(long)(C + c) * Ho * Wo + p] = (half_t)(sm * (1.f / 9.f));
}

// Modulated deformable-conv sampling (one frame): builds im2col tensor
// s[(g*8+c)*9+k, y, x] = mask * bilinear(x_feat[g*8+c], y+ky+dy, x+kx+dx)
// o: 216 channels = [dy(72) | dx(72) | mask(72)], thread per (g,k,y,x).
__global__ void dcn_sample_kernel(const half_t* __restrict__ x, const half_t* __restrict__ o,
                                  half_t* __restrict__ s, int H, int W)
{
  int HW = H * W;
  int total = 72 * HW;
  int i = blockIdx.x * 256 + threadIdx.x;
  if (i >= total) return;
  int gk = i / HW;
  int p  = i - gk * HW;
  int g  = gk / 9, k = gk - g * 9;
  int yy = p / W, xx = p - yy * W;
  float dy = (float)o[(long)gk * HW + p];
  float dx = (float)o[(long)(72 + gk) * HW + p];
  float mk = dsigmoid((float)o[(long)(144 + gk) * HW + p]);
  float py = (float)yy + (float)(k / 3 - 1) + dy;
  float px = (float)xx + (float)(k % 3 - 1) + dx;
  float y0f = floorf(py), x0f = floorf(px);
  int y0 = (int)y0f, x0 = (int)x0f;
  float ty = py - y0f, tx = px - x0f;
  int y1 = y0 + 1, x1 = x0 + 1;
  bool by0 = (y0 >= 0 && y0 < H), by1 = (y1 >= 0 && y1 < H);
  bool bx0 = (x0 >= 0 && x0 < W), bx1 = (x1 >= 0 && x1 < W);
  float w00 = (by0 && bx0) ? (1.f - ty) * (1.f - tx) : 0.f;
  float w01 = (by0 && bx1) ? (1.f - ty) * tx : 0.f;
  float w10 = (by1 && bx0) ? ty * (1.f - tx) : 0.f;
  float w11 = (by1 && bx1) ? ty * tx : 0.f;
  int y0c = y0 < 0 ? 0 : (y0 > H - 1 ? H - 1 : y0);
  int y1c = y1 < 0 ? 0 : (y1 > H - 1 ? H - 1 : y1);
  int x0c = x0 < 0 ? 0 : (x0 > W - 1 ? W - 1 : x0);
  int x1c = x1 < 0 ? 0 : (x1 > W - 1 ? W - 1 : x1);
#pragma unroll
  for (int c = 0; c < 8; ++c) {
    const half_t* ch = x + (long)(g * 8 + c) * HW;
    float v = w00 * (float)ch[y0c * W + x0c] + w01 * (float)ch[y0c * W + x1c] +
              w10 * (float)ch[y1c * W + x0c] + w11 * (float)ch[y1c * W + x1c];
    s[(long)((g * 8 + c) * 9 + k) * HW + p] = (half_t)(v * mk);
  }
}

// TSA temporal attention: prob = sigmoid(<emb[n,t], emb_ref[n]>_c), af[n, t*64+c] = a * prob
__global__ void tsa_prob_af_kernel(const half_t* __restrict__ emb,
                                   const half_t* __restrict__ embref,
                                   const half_t* __restrict__ a,
                                   half_t* __restrict__ af, int HW)
{
  int total = 10 * HW;
  int i = blockIdx.x * 256 + threadIdx.x;
  if (i >= total) return;
  int nt = i / HW;
  int p  = i - nt * HW;
  int n  = nt / 5, t = nt - n * 5;
  const half_t* e  = emb + (long)nt * 64 * HW + p;
  const half_t* er = embref + (long)n * 64 * HW + p;
  float dot = 0.f;
#pragma unroll 8
  for (int c = 0; c < 64; ++c)
    dot += (float)e[(long)c * HW] * (float)er[(long)c * HW];
  float prob = dsigmoid(dot);
  const half_t* av = a + (long)nt * 64 * HW + p;
  half_t* op = af + ((long)n * 320 + t * 64) * HW + p;
#pragma unroll 8
  for (int c = 0; c < 64; ++c)
    op[(long)c * HW] = (half_t)((float)av[(long)c * HW] * prob);
}

__global__ void final_combine_kernel(const half_t* __restrict__ feat,
                                     const half_t* __restrict__ attn,
                                     const half_t* __restrict__ addv,
                                     float* __restrict__ out, int n)
{
  int i = blockIdx.x * 256 + threadIdx.x;
  if (i >= n) return;
  float s = dsigmoid((float)attn[i]);
  out[i] = (float)feat[i] * s * 2.f + (float)addv[i];
}

// ---------------------------------------------------------------------------
// Host side
// ---------------------------------------------------------------------------
struct WSpec { int cout, cin, ks; };

// sorted(shapes) order from the reference; d_in[1+2i]=weight f32, d_in[2+2i]=bias f32
static const WSpec g_wspec[48] = {
  {64,64,3},   // cas_dcn
  {216,64,3},  // cas_dcnoff
  {64,128,3},  // cas_oc1
  {64,64,3},   // cas_oc2
  {64,3,3},    // conv_first
  {64,320,1},  // fusion
  {64,64,3},   // l1_dcn
  {216,64,3},  // l1_dcnoff
  {64,128,3},  // l1_fc
  {64,128,3},  // l1_oc1
  {64,128,3},  // l1_oc2
  {64,64,3},   // l1_oc3
  {64,64,3},   // l2_dcn
  {216,64,3},  // l2_dcnoff
  {64,128,3},  // l2_fc
  {64,128,3},  // l2_oc1
  {64,128,3},  // l2_oc2
  {64,64,3},   // l2_oc3
  {64,64,3},   // l2c1
  {64,64,3},   // l2c2
  {64,64,3},   // l3_dcn
  {216,64,3},  // l3_dcnoff
  {64,128,3},  // l3_oc1
  {64,64,3},   // l3_oc2
  {64,64,3},   // l3c1
  {64,64,3},   // l3c2
  {64,64,3},{64,64,3},{64,64,3},{64,64,3},{64,64,3},
  {64,64,3},{64,64,3},{64,64,3},{64,64,3},{64,64,3},   // rb0a..rb4b
  {64,320,1},  // sa1
  {64,128,1},  // sa2
  {64,64,3},   // sa3
  {64,64,1},   // sa4
  {64,64,3},   // sa5
  {64,64,1},   // saa1
  {64,64,1},   // saa2
  {64,64,1},   // sal1
  {64,128,3},  // sal2
  {64,64,3},   // sal3
  {64,64,3},   // ta1
  {64,64,3},   // ta2
};
enum { W_CAS_DCN = 0, W_CAS_DCNOFF, W_CAS_OC1, W_CAS_OC2, W_CONV_FIRST, W_FUSION,
       W_L1_DCN, W_L1_DCNOFF, W_L1_FC, W_L1_OC1, W_L1_OC2, W_L1_OC3,
       W_L2_DCN, W_L2_DCNOFF, W_L2_FC, W_L2_OC1, W_L2_OC2, W_L2_OC3,
       W_L2C1, W_L2C2,
       W_L3_DCN, W_L3_DCNOFF, W_L3_OC1, W_L3_OC2, W_L3C1, W_L3C2,
       W_RB0A, W_RB0B, W_RB1A, W_RB1B, W_RB2A, W_RB2B, W_RB3A, W_RB3B, W_RB4A, W_RB4B,
       W_SA1, W_SA2, W_SA3, W_SA4, W_SA5, W_SAA1, W_SAA2,
       W_SAL1, W_SAL2, W_SAL3, W_TA1, W_TA2, W_COUNT };

static void run_conv(hipStream_t st, const half_t* in, const half_t* w, const float* b,
                     half_t* out, int Cin, int H, int W, int Cout,
                     int stride, int ks, int act, int nImg, long inStride)
{
  int pad = (ks == 3) ? 1 : 0;
  int Ho = (H + 2 * pad - ks) / stride + 1;
  int Wo = (W + 2 * pad - ks) / stride + 1;
  int tilesX = (Wo + 31) / 32;
  int CoutPad = (Cout + 15) & ~15;
  int coutBlocks = (CoutPad + 63) / 64;     // 1 (Cout<=64) or 4 (Cout=216)
  int zShift = (coutBlocks > 1) ? 2 : 0;
  int zMask  = (coutBlocks > 1) ? 3 : 0;
  dim3 grid(tilesX, Ho, nImg << zShift);
  long outStride = (long)Cout * Ho * Wo;
  if (ks == 3 && stride == 1)
    conv_wmma_kernel<3, 1><<<grid, 128, 0, st>>>(in, w, b, out, Cin, H, W, Cout, Ho, Wo,
                                                 act, zShift, zMask, inStride, outStride);
  else if (ks == 3)
    conv_wmma_kernel<3, 2><<<grid, 128, 0, st>>>(in, w, b, out, Cin, H, W, Cout, Ho, Wo,
                                                 act, zShift, zMask, inStride, outStride);
  else
    conv_wmma_kernel<1, 1><<<grid, 128, 0, st>>>(in, w, b, out, Cin, H, W, Cout, Ho, Wo,
                                                 act, zShift, zMask, inStride, outStride);
}

extern "C" void kernel_launch(void* const* d_in, const int* in_sizes, int n_in,
                              void* d_out, int out_size, void* d_ws, size_t ws_size,
                              hipStream_t stream)
{
  (void)in_sizes; (void)n_in; (void)out_size; (void)ws_size;
  const int H1 = 96, W1 = 160, H2 = 48, W2 = 80, H3 = 24, W3 = 40;
  const int HW1 = H1 * W1, HW2 = H2 * W2, HW3 = H3 * W3;
  const int NI = 10;
  const long F1 = 64L * HW1, F2 = 64L * HW2, F3 = 64L * HW3;

  char* base = (char*)d_ws;
  size_t cur = 0;
  auto alloc = [&](long elems) -> half_t* {
    half_t* p = (half_t*)(base + cur);
    cur = (cur + (size_t)elems * sizeof(half_t) + 255) & ~(size_t)255;
    return p;
  };

  // ---- pack weights f32 -> f16 [CoutPad][Kpad] ----
  half_t* wpk[W_COUNT];
  for (int i = 0; i < W_COUNT; ++i) {
    const WSpec& s = g_wspec[i];
    int K = s.cin * s.ks * s.ks;
    int Kpad = (K + 31) & ~31;
    int CoutPad = (s.cout + 15) & ~15;
    int total = CoutPad * Kpad;
    wpk[i] = alloc(total);
    pack_weight_kernel<<<(total + 255) / 256, 256, 0, stream>>>(
        (const float*)d_in[1 + 2 * i], wpk[i], s.cout, K, Kpad, total);
  }
  auto BIAS = [&](int wi) -> const float* { return (const float*)d_in[2 + 2 * wi]; };
  auto CONV = [&](const half_t* in, int wi, half_t* out, int Cin, int H, int W,
                  int stride, int ks, int act, int nImg, long inStride) {
    run_conv(stream, in, wpk[wi], BIAS(wi), out, Cin, H, W, g_wspec[wi].cout,
             stride, ks, act, nImg, inStride);
  };
  auto CONVD = [&](const half_t* in, int wi, half_t* out, int Cin, int H, int W,
                   int stride, int ks, int act, int nImg) {
    CONV(in, wi, out, Cin, H, W, stride, ks, act, nImg, (long)Cin * H * W);
  };
  auto copych = [&](const half_t* src, half_t* dst, int C, int HW, int nImg,
                    int dstC, int off, int ctr) {
    int n = nImg * C * HW;
    copy_ch_kernel<<<(n + 255) / 256, 256, 0, stream>>>(src, dst, C, HW, nImg, dstC, off, ctr);
  };

  // ---- workspace buffers (f16) ----
  half_t* xh     = alloc((long)NI * 3 * HW1);
  half_t* l1     = alloc(NI * F1);
  half_t* ta     = alloc(NI * F1);
  half_t* tb     = alloc(NI * F1);
  half_t* l2a    = alloc(NI * F2);
  half_t* l2b    = alloc(NI * F2);
  half_t* l3a    = alloc(NI * F3);
  half_t* l3b    = alloc(NI * F3);
  half_t* cat    = alloc(NI * 2 * F1);
  half_t* off3a  = alloc(NI * F3);
  half_t* off3b  = alloc(NI * F3);
  half_t* off2a  = alloc(NI * F2);
  half_t* off2b  = alloc(NI * F2);
  half_t* off1a  = alloc(NI * F1);
  half_t* off1b  = alloc(NI * F1);
  half_t* upo2   = alloc(NI * F2);
  half_t* upo1   = alloc(NI * F1);
  half_t* upf2   = alloc(NI * F2);
  half_t* upf1   = alloc(NI * F1);
  half_t* feat3  = alloc(NI * F3);
  half_t* f2tmp  = alloc(NI * F2);
  half_t* feat2  = alloc(NI * F2);
  half_t* f1tmp  = alloc(NI * F1);
  half_t* feat1  = alloc(NI * F1);
  half_t* featc  = alloc(NI * F1);
  half_t* obuf   = alloc((long)NI * 216 * HW1);
  half_t* sbuf   = alloc(576L * HW1);
  half_t* embref = alloc(2 * F1);
  half_t* emb    = alloc(NI * F1);
  half_t* af     = alloc(2L * 5 * F1);
  half_t* fts    = alloc(2 * F1);
  half_t* attn1  = alloc(2 * F1);
  half_t* atp    = alloc(2L * 2 * F2);
  half_t* attn2  = alloc(2 * F2);
  half_t* al1    = alloc(2 * F2);
  half_t* alp    = alloc(2L * 2 * F3);
  half_t* al2    = alloc(2 * F3);
  half_t* al3    = alloc(2 * F3);
  half_t* alup   = alloc(2 * F2);
  half_t* attn3  = alloc(2 * F2);
  half_t* attn4  = alloc(2 * F2);
  half_t* attnup = alloc(2 * F1);
  half_t* attn5  = alloc(2 * F1);
  half_t* aa1    = alloc(2 * F1);
  half_t* aadd   = alloc(2 * F1);

  // ================= feature extraction =================
  {
    int n = NI * 3 * HW1;
    cvt_f32_f16_kernel<<<(n + 255) / 256, 256, 0, stream>>>((const float*)d_in[0], xh, n);
  }
  CONVD(xh, W_CONV_FIRST, l1, 3, H1, W1, 1, 3, ACT_LRELU, NI);
  for (int r = 0; r < 5; ++r) {                 // ResidualBlockNoBN x5
    CONVD(l1, W_RB0A + 2 * r, ta, 64, H1, W1, 1, 3, ACT_RELU, NI);
    CONVD(ta, W_RB0B + 2 * r, tb, 64, H1, W1, 1, 3, ACT_NONE, NI);
    int n = (int)(NI * F1);
    add_f16_kernel<<<(n + 255) / 256, 256, 0, stream>>>(l1, tb, n);
  }
  CONVD(l1,  W_L2C1, l2a, 64, H1, W1, 2, 3, ACT_LRELU, NI);
  CONVD(l2a, W_L2C2, l2b, 64, H2, W2, 1, 3, ACT_LRELU, NI);
  CONVD(l2b, W_L3C1, l3a, 64, H2, W2, 2, 3, ACT_LRELU, NI);
  CONVD(l3a, W_L3C2, l3b, 64, H3, W3, 1, 3, ACT_LRELU, NI);

  // ================= PCD align: level 3 =================
  copych(l3b, cat, 64, HW3, NI, 128, 0, 0);
  copych(l3b, cat, 64, HW3, NI, 128, 64, 1);
  CONVD(cat,   W_L3_OC1,    off3a, 128, H3, W3, 1, 3, ACT_LRELU, NI);
  CONVD(off3a, W_L3_OC2,    off3b, 64,  H3, W3, 1, 3, ACT_LRELU, NI);
  CONVD(off3b, W_L3_DCNOFF, obuf,  64,  H3, W3, 1, 3, ACT_NONE,  NI);
  for (int f = 0; f < NI; ++f) {
    int n = 72 * HW3;
    dcn_sample_kernel<<<(n + 255) / 256, 256, 0, stream>>>(
        l3b + f * F3, obuf + (long)f * 216 * HW3, sbuf, H3, W3);
    CONVD(sbuf, W_L3_DCN, feat3 + f * F3, 576, H3, W3, 1, 1, ACT_LRELU, 1);
  }
  {
    int n = NI * 64 * HW2;
    up2_kernel<<<(n + 255) / 256, 256, 0, stream>>>(off3b, upo2, 64, H3, W3, NI, 2.0f);
    up2_kernel<<<(n + 255) / 256, 256, 0, stream>>>(feat3, upf2, 64, H3, W3, NI, 1.0f);
  }

  // ================= PCD align: level 2 =================
  copych(l2b, cat, 64, HW2, NI, 128, 0, 0);
  copych(l2b, cat, 64, HW2, NI, 128, 64, 1);
  CONVD(cat, W_L2_OC1, off2a, 128, H2, W2, 1, 3, ACT_LRELU, NI);
  copych(off2a, cat, 64, HW2, NI, 128, 0, 0);
  copych(upo2,  cat, 64, HW2, NI, 128, 64, 0);
  CONVD(cat,   W_L2_OC2,    off2a, 128, H2, W2, 1, 3, ACT_LRELU, NI);
  CONVD(off2a, W_L2_OC3,    off2b, 64,  H2, W2, 1, 3, ACT_LRELU, NI);
  CONVD(off2b, W_L2_DCNOFF, obuf,  64,  H2, W2, 1, 3, ACT_NONE,  NI);
  for (int f = 0; f < NI; ++f) {
    int n = 72 * HW2;
    dcn_sample_kernel<<<(n + 255) / 256, 256, 0, stream>>>(
        l2b + f * F2, obuf + (long)f * 216 * HW2, sbuf, H2, W2);
    CONVD(sbuf, W_L2_DCN, f2tmp + f * F2, 576, H2, W2, 1, 1, ACT_NONE, 1);
  }
  copych(f2tmp, cat, 64, HW2, NI, 128, 0, 0);
  copych(upf2,  cat, 64, HW2, NI, 128, 64, 0);
  CONVD(cat, W_L2_FC, feat2, 128, H2, W2, 1, 3, ACT_LRELU, NI);
  {
    int n = NI * 64 * HW1;
    up2_kernel<<<(n + 255) / 256, 256, 0, stream>>>(off2b, upo1, 64, H2, W2, NI, 2.0f);
    up2_kernel<<<(n + 255) / 256, 256, 0, stream>>>(feat2, upf1, 64, H2, W2, NI, 1.0f);
  }

  // ================= PCD align: level 1 =================
  copych(l1, cat, 64, HW1, NI, 128, 0, 0);
  copych(l1, cat, 64, HW1, NI, 128, 64, 1);
  CONVD(cat, W_L1_OC1, off1a, 128, H1, W1, 1, 3, ACT_LRELU, NI);
  copych(off1a, cat, 64, HW1, NI, 128, 0, 0);
  copych(upo1,  cat, 64, HW1, NI, 128, 64, 0);
  CONVD(cat,   W_L1_OC2,    off1a, 128, H1, W1, 1, 3, ACT_LRELU, NI);
  CONVD(off1a, W_L1_OC3,    off1b, 64,  H1, W1, 1, 3, ACT_LRELU, NI);
  CONVD(off1b, W_L1_DCNOFF, obuf,  64,  H1, W1, 1, 3, ACT_NONE,  NI);
  for (int f = 0; f < NI; ++f) {
    int n = 72 * HW1;
    dcn_sample_kernel<<<(n + 255) / 256, 256, 0, stream>>>(
        l1 + f * F1, obuf + (long)f * 216 * HW1, sbuf, H1, W1);
    CONVD(sbuf, W_L1_DCN, f1tmp + f * F1, 576, H1, W1, 1, 1, ACT_NONE, 1);
  }
  copych(f1tmp, cat, 64, HW1, NI, 128, 0, 0);
  copych(upf1,  cat, 64, HW1, NI, 128, 64, 0);
  CONVD(cat, W_L1_FC, feat1, 128, H1, W1, 1, 3, ACT_NONE, NI);

  // ================= PCD cascade =================
  copych(feat1, cat, 64, HW1, NI, 128, 0, 0);
  copych(l1,    cat, 64, HW1, NI, 128, 64, 1);
  CONVD(cat,   W_CAS_OC1,    off1a, 128, H1, W1, 1, 3, ACT_LRELU, NI);
  CONVD(off1a, W_CAS_OC2,    off1b, 64,  H1, W1, 1, 3, ACT_LRELU, NI);
  CONVD(off1b, W_CAS_DCNOFF, obuf,  64,  H1, W1, 1, 3, ACT_NONE,  NI);
  for (int f = 0; f < NI; ++f) {
    int n = 72 * HW1;
    dcn_sample_kernel<<<(n + 255) / 256, 256, 0, stream>>>(
        feat1 + f * F1, obuf + (long)f * 216 * HW1, sbuf, H1, W1);
    CONVD(sbuf, W_CAS_DCN, featc + f * F1, 576, H1, W1, 1, 1, ACT_LRELU, 1);
  }

  // ================= TSA fusion =================
  // emb_ref over the two center frames (2 and 7): strided nImg=2 view
  CONV(featc + 2 * F1, W_TA1, embref, 64, H1, W1, 1, 3, ACT_NONE, 2, 5 * F1);
  CONVD(featc, W_TA2, emb, 64, H1, W1, 1, 3, ACT_NONE, NI);
  {
    int n = NI * HW1;
    tsa_prob_af_kernel<<<(n + 255) / 256, 256, 0, stream>>>(emb, embref, featc, af, HW1);
  }
  CONVD(af, W_FUSION, fts,   320, H1, W1, 1, 1, ACT_LRELU, 2);
  CONVD(af, W_SA1,    attn1, 320, H1, W1, 1, 1, ACT_LRELU, 2);
  {
    int n = 2 * 64 * HW2;
    pool_dual_kernel<<<(n + 255) / 256, 256, 0, stream>>>(attn1, atp, 64, H1, W1, 2);
  }
  CONVD(atp,   W_SA2,  attn2, 128, H2, W2, 1, 1, ACT_LRELU, 2);
  CONVD(attn2, W_SAL1, al1,   64,  H2, W2, 1, 1, ACT_LRELU, 2);
  {
    int n = 2 * 64 * HW3;
    pool_dual_kernel<<<(n + 255) / 256, 256, 0, stream>>>(al1, alp, 64, H2, W2, 2);
  }
  CONVD(alp, W_SAL2, al2, 128, H3, W3, 1, 3, ACT_LRELU, 2);
  CONVD(al2, W_SAL3, al3, 64,  H3, W3, 1, 3, ACT_LRELU, 2);
  {
    int n = 2 * 64 * HW2;
    up2_kernel<<<(n + 255) / 256, 256, 0, stream>>>(al3, alup, 64, H3, W3, 2, 1.0f);
  }
  CONVD(attn2, W_SA3, attn3, 64, H2, W2, 1, 3, ACT_LRELU, 2);
  {
    int n = (int)(2 * F2);
    add_f16_kernel<<<(n + 255) / 256, 256, 0, stream>>>(attn3, alup, n);
  }
  CONVD(attn3, W_SA4, attn4, 64, H2, W2, 1, 1, ACT_LRELU, 2);
  {
    int n = 2 * 64 * HW1;
    up2_kernel<<<(n + 255) / 256, 256, 0, stream>>>(attn4, attnup, 64, H2, W2, 2, 1.0f);
  }
  CONVD(attnup, W_SA5,  attn5, 64, H1, W1, 1, 3, ACT_NONE,  2);
  CONVD(attn5,  W_SAA1, aa1,   64, H1, W1, 1, 1, ACT_LRELU, 2);
  CONVD(aa1,    W_SAA2, aadd,  64, H1, W1, 1, 1, ACT_NONE,  2);
  {
    int n = (int)(2 * F1);
    final_combine_kernel<<<(n + 255) / 256, 256, 0, stream>>>(fts, attn5, aadd,
                                                              (float*)d_out, n);
  }
}
